// MiniCPMV_ViTWindowAttentionMerger_36369783062534
// MI455X (gfx1250) — matrix operand order repaired and checked by reading, not compiled, not run on hardware
//
#include <hip/hip_runtime.h>
#include <hip/hip_bf16.h>

#define DEV __device__ __forceinline__

typedef __attribute__((ext_vector_type(16))) __bf16 v16bf;
typedef __attribute__((ext_vector_type(8)))  __bf16 v8bf;
typedef __attribute__((ext_vector_type(8)))  float  v8f;

// Fixed problem shapes (from setup_inputs)
constexpr int Bc = 4, Hc = 64, Wc = 64, Dc = 1152, NHc = 16, HDc = 72;
constexpr int Tc   = Bc * Hc * Wc;   // 16384 tokens
constexpr int NWIN = Tc / 4;         // 4096 windows / merged tokens
constexpr int D3   = 3 * Dc;         // 3456
constexpr int D4   = 4 * Dc;         // 4608
constexpr int I4   = 4 * 4304;       // 17216

// ---------- scalar bf16 helpers (RNE) ----------
DEV __bf16 f2bf(float f) {
  unsigned u = __builtin_bit_cast(unsigned, f);
  unsigned r = u + 0x7FFFu + ((u >> 16) & 1u);
  unsigned short s = (unsigned short)(r >> 16);
  return __builtin_bit_cast(__bf16, s);
}
DEV float bf2f(__bf16 h) {
  unsigned short s = __builtin_bit_cast(unsigned short, h);
  unsigned u = ((unsigned)s) << 16;
  return __builtin_bit_cast(float, u);
}
DEV v8f zero8() {
  v8f z;
#pragma unroll
  for (int i = 0; i < 8; ++i) z[i] = 0.0f;
  return z;
}

DEV v8f wmma_bf16(v16bf a, v16bf b, v8f c) {
  // D = A(16x32,bf16) x B(32x16,bf16) + C(16x16,f32)
  return __builtin_amdgcn_wmma_f32_16x16x32_bf16(false, a, false, b,
                                                 (short)0, c, false, false);
}

// window-ordered row r (in [0,16384)) -> original token index t
DEV int token_of(int win, int pos) {
  int b   = win >> 10;        // nh*nw = 1024
  int rem = win & 1023;
  int y   = rem >> 5;         // nw = 32
  int x   = rem & 31;
  return b * 4096 + y * 128 + x * 2 + (pos >> 1) * 64 + (pos & 1);
}
DEV int win_r2t(int r) { return token_of(r >> 2, r & 3); }

// ---------- fp32 -> bf16 weight conversion ----------
__global__ void k_f32_to_bf16(const float* __restrict__ in,
                              __bf16* __restrict__ out, long n) {
  long i = (long)blockIdx.x * 256 + threadIdx.x;
  if (i < n) out[i] = f2bf(in[i]);
}

// ---------- LN1 + window reorder -> bf16 A matrix ----------
__global__ void k_ln1(const float* __restrict__ x, const float* __restrict__ g,
                      const float* __restrict__ b, __bf16* __restrict__ Aw) {
  __shared__ float ssum[256], ssq[256];
  const int r = blockIdx.x;
  const int t = win_r2t(r);
  const float* xp = x + (long)t * Dc;
  float sum = 0.f, sq = 0.f;
  for (int d = threadIdx.x; d < Dc; d += 256) { float v = xp[d]; sum += v; sq += v * v; }
  ssum[threadIdx.x] = sum; ssq[threadIdx.x] = sq; __syncthreads();
  for (int s = 128; s > 0; s >>= 1) {
    if (threadIdx.x < (unsigned)s) { ssum[threadIdx.x] += ssum[threadIdx.x + s];
                                     ssq[threadIdx.x]  += ssq[threadIdx.x + s]; }
    __syncthreads();
  }
  const float mu   = ssum[0] * (1.0f / Dc);
  const float var  = ssq[0] * (1.0f / Dc) - mu * mu;
  const float rstd = rsqrtf(var + 1e-6f);
  __bf16* op = Aw + (long)r * Dc;
  for (int d = threadIdx.x; d < Dc; d += 256)
    op[d] = f2bf((xp[d] - mu) * rstd * g[d] + b[d]);
}

// ---------- per-window attention (tiny 4x4 per head) ----------
__global__ void k_attn(const __bf16* __restrict__ qkv, __bf16* __restrict__ O) {
  int tid = blockIdx.x * 256 + threadIdx.x;
  if (tid >= NWIN * NHc * 4) return;
  const int qi  = tid & 3;
  const int h   = (tid >> 2) & 15;
  const int win = tid >> 6;
  const long rs = D3;
  const __bf16* qp    = qkv + (long)(win * 4 + qi) * rs + h * HDc;
  const __bf16* kbase = qkv + (long)(win * 4) * rs + Dc + h * HDc;
  const __bf16* vbase = qkv + (long)(win * 4) * rs + 2 * Dc + h * HDc;
  const float scale = 0.11785113019775793f;  // 1/sqrt(72)
  float s[4];
#pragma unroll
  for (int j = 0; j < 4; ++j) {
    float acc = 0.f;
    for (int d = 0; d < HDc; ++d)
      acc += bf2f(qp[d]) * bf2f(kbase[(long)j * rs + d]);
    s[j] = acc * scale;
  }
  float mx = fmaxf(fmaxf(s[0], s[1]), fmaxf(s[2], s[3]));
  float e0 = expf(s[0] - mx), e1 = expf(s[1] - mx),
        e2 = expf(s[2] - mx), e3 = expf(s[3] - mx);
  float inv = 1.0f / (e0 + e1 + e2 + e3);
  float p0 = e0 * inv, p1 = e1 * inv, p2 = e2 * inv, p3 = e3 * inv;
  __bf16* op = O + (long)(win * 4 + qi) * Dc + h * HDc;
  for (int d = 0; d < HDc; ++d) {
    float acc = p0 * bf2f(vbase[d]) + p1 * bf2f(vbase[rs + d]) +
                p2 * bf2f(vbase[2 * rs + d]) + p3 * bf2f(vbase[3 * rs + d]);
    op[d] = f2bf(acc);
  }
}

// ---------- stage-2 LN over folded 2x2 windows -> bf16 [4096 x 4608] ----------
__global__ void k_ln2(const float* __restrict__ X2, const float* __restrict__ g,
                      const float* __restrict__ b, __bf16* __restrict__ Mo) {
  __shared__ float ssum[256], ssq[256];
  const int m = blockIdx.x;
  const int base = token_of(m, 0);
  float sum = 0.f, sq = 0.f;
  for (int j = threadIdx.x; j < D4; j += 256) {
    int pos = j / Dc, d = j - pos * Dc;
    int t = base + (pos >> 1) * 64 + (pos & 1);
    float v = X2[(long)t * Dc + d];
    sum += v; sq += v * v;
  }
  ssum[threadIdx.x] = sum; ssq[threadIdx.x] = sq; __syncthreads();
  for (int s = 128; s > 0; s >>= 1) {
    if (threadIdx.x < (unsigned)s) { ssum[threadIdx.x] += ssum[threadIdx.x + s];
                                     ssq[threadIdx.x]  += ssq[threadIdx.x + s]; }
    __syncthreads();
  }
  const float mu   = ssum[0] * (1.0f / D4);
  const float var  = ssq[0] * (1.0f / D4) - mu * mu;
  const float rstd = rsqrtf(var + 1e-6f);
  __bf16* op = Mo + (long)m * D4;
  for (int j = threadIdx.x; j < D4; j += 256) {
    int pos = j / Dc, d = j - pos * Dc;
    int t = base + (pos >> 1) * 64 + (pos & 1);
    float v = X2[(long)t * Dc + d];
    op[j] = f2bf((v - mu) * rstd * g[j] + b[j]);
  }
}

// ---------- mean-pool residual [4096 x 1152] ----------
__global__ void k_res(const float* __restrict__ X2, float* __restrict__ res) {
  long idx = (long)blockIdx.x * 256 + threadIdx.x;
  if (idx >= (long)NWIN * Dc) return;
  int m = (int)(idx / Dc);
  int d = (int)(idx - (long)m * Dc);
  int base = token_of(m, 0);
  float s = X2[(long)base * Dc + d] + X2[(long)(base + 1) * Dc + d] +
            X2[(long)(base + 64) * Dc + d] + X2[(long)(base + 65) * Dc + d];
  res[idx] = 0.25f * s;
}

// ---------- WMMA bf16 GEMM: C[M,N] = A[M,K] * W[N,K]^T, fused epilogues ----------
// Block tile 256(M) x 128(N), 512 threads = 16 waves (4x4), wave tile 64x32.
// N may be ragged (FC1: 17216 = 134.5*128): B-row pointers clamp, stores guard.
// EPI 0: +bias        -> bf16 store           (QKV)
// EPI 1: +bias +x[t]  -> f32 store, token map (attn out-proj + residual)
// EPI 2: +bias, GELU  -> bf16 store           (FC1)
// EPI 3: +bias +res   -> f32 store            (FC2, final output)
template <int EPI>
__global__ __launch_bounds__(512, 1) void k_gemm(
    const __bf16* __restrict__ A, const __bf16* __restrict__ Wt,
    void* __restrict__ Cout, const float* __restrict__ bias,
    const float* __restrict__ extra, int M, int N, int K) {
  const int lane = threadIdx.x & 31;
  const int wave = threadIdx.x >> 5;
  const int wr = wave >> 2, wc = wave & 3;          // 4 x 4 waves
  const int m_base = blockIdx.y * 256 + wr * 64;    // wave: 64 rows
  const int n_base = blockIdx.x * 128 + wc * 32;    // wave: 32 cols
  const int rl = lane & 15, kh = lane >> 4;

  v8f c[4][2];
#pragma unroll
  for (int mt = 0; mt < 4; ++mt) {
    c[mt][0] = zero8(); c[mt][1] = zero8();
  }

  const __bf16* ap[4];
#pragma unroll
  for (int mt = 0; mt < 4; ++mt)
    ap[mt] = A + (long)(m_base + mt * 16 + rl) * K;
  const __bf16* bp[2];
#pragma unroll
  for (int nt = 0; nt < 2; ++nt) {
    int nr = n_base + nt * 16 + rl;
    if (nr > N - 1) nr = N - 1;                     // clamp for ragged N
    bp[nt] = Wt + (long)nr * K;
  }

  for (int k0 = 0; k0 < K; k0 += 32) {
    v16bf b0, b1;
    b0 = *(const v16bf*)(bp[0] + k0 + kh * 16);
    b1 = *(const v16bf*)(bp[1] + k0 + kh * 16);
    __builtin_prefetch((const void*)(bp[0] + k0 + 64 + kh * 16), 0, 1);
    __builtin_prefetch((const void*)(bp[1] + k0 + 64 + kh * 16), 0, 1);
#pragma unroll
    for (int mt = 0; mt < 4; ++mt) {
      v16bf a;
      {
        v8bf lo = *(const v8bf*)(ap[mt] + k0 + kh * 16);
        v8bf hi = *(const v8bf*)(ap[mt] + k0 + kh * 16 + 8);
#pragma unroll
        for (int i = 0; i < 8; ++i) { a[i] = lo[i]; a[8 + i] = hi[i]; }
      }
      c[mt][0] = wmma_bf16(a, b0, c[mt][0]);
      c[mt][1] = wmma_bf16(a, b1, c[mt][1]);
    }
  }

#pragma unroll
  for (int mt = 0; mt < 4; ++mt) {
#pragma unroll
    for (int nt = 0; nt < 2; ++nt) {
      const int col = n_base + nt * 16 + rl;
      if (col >= N) continue;
      const float bv = bias[col];
      const int row0 = m_base + mt * 16 + kh * 8;
#pragma unroll
      for (int v = 0; v < 8; ++v) {
        const int row = row0 + v;
        float val = c[mt][nt][v] + bv;
        if constexpr (EPI == 0) {
          ((__bf16*)Cout)[(long)row * N + col] = f2bf(val);
        } else if constexpr (EPI == 1) {
          const int t = win_r2t(row);
          const long idx = (long)t * Dc + col;
          ((float*)Cout)[idx] = val + extra[idx];
        } else if constexpr (EPI == 2) {
          float x3 = val * val * val;
          float gv = 0.5f * val *
                     (1.0f + tanhf(0.7978845608028654f * (val + 0.044715f * x3)));
          ((__bf16*)Cout)[(long)row * N + col] = f2bf(gv);
        } else {
          const long idx = (long)row * Dc + col;
          ((float*)Cout)[idx] = val + extra[idx];
        }
      }
    }
  }
}

extern "C" void kernel_launch(void* const* d_in, const int* in_sizes, int n_in,
                              void* d_out, int out_size, void* d_ws, size_t ws_size,
                              hipStream_t stream) {
  const float* x    = (const float*)d_in[0];
  const float* ln1g = (const float*)d_in[1];
  const float* ln1b = (const float*)d_in[2];
  const float* wqkv = (const float*)d_in[3];
  const float* bqkv = (const float*)d_in[4];
  const float* wo   = (const float*)d_in[5];
  const float* bo   = (const float*)d_in[6];
  const float* preg = (const float*)d_in[7];
  const float* preb = (const float*)d_in[8];
  const float* w1   = (const float*)d_in[9];
  const float* b1   = (const float*)d_in[10];
  const float* w2   = (const float*)d_in[11];
  const float* b2   = (const float*)d_in[12];
  (void)in_sizes; (void)n_in; (void)out_size; (void)ws_size;

  char* ws = (char*)d_ws;
  size_t off = 0;
  auto alloc = [&](size_t bytes) -> char* {
    char* p = ws + off; off = (off + bytes + 255) & ~(size_t)255; return p;
  };
  __bf16* Wqkv_bf = (__bf16*)alloc((size_t)D3 * Dc * 2);
  __bf16* Wo_bf   = (__bf16*)alloc((size_t)Dc * Dc * 2);
  __bf16* W1_bf   = (__bf16*)alloc((size_t)I4 * D4 * 2);
  __bf16* W2_bf   = (__bf16*)alloc((size_t)Dc * I4 * 2);
  float*  X2      = (float*)alloc((size_t)Tc * Dc * 4);
  float*  RES     = (float*)alloc((size_t)NWIN * Dc * 4);
  char*   BufA    = alloc((size_t)Tc * Dc * 2);   // Aw -> O -> Mln (lifetimes disjoint)
  size_t qb = (size_t)Tc * D3 * 2, hb = (size_t)NWIN * I4 * 2;
  char*   BufB    = alloc(qb > hb ? qb : hb);     // QKV -> H1 (lifetimes disjoint)

  __bf16* Aw  = (__bf16*)BufA;   // LN1'd activations, window order [16384 x 1152]
  __bf16* QKV = (__bf16*)BufB;   // [16384 x 3456]
  __bf16* O   = (__bf16*)BufA;   // attention out [16384 x 1152]
  __bf16* Mln = (__bf16*)BufA;   // stage-2 LN    [4096 x 4608]
  __bf16* H1  = (__bf16*)BufB;   // gelu(fc1)     [4096 x 17216]

  dim3 blk(256), gblk(512);
  auto cvt = [&](const float* src, __bf16* dst, long n) {
    k_f32_to_bf16<<<dim3((unsigned)((n + 255) / 256)), blk, 0, stream>>>(src, dst, n);
  };
  cvt(wqkv, Wqkv_bf, (long)D3 * Dc);
  cvt(wo,   Wo_bf,   (long)Dc * Dc);
  cvt(w1,   W1_bf,   (long)I4 * D4);
  cvt(w2,   W2_bf,   (long)Dc * I4);

  // Stage 1
  k_ln1<<<dim3(Tc), blk, 0, stream>>>(x, ln1g, ln1b, Aw);
  k_gemm<0><<<dim3(D3 / 128, Tc / 256), gblk, 0, stream>>>(
      Aw, Wqkv_bf, (void*)QKV, bqkv, nullptr, Tc, D3, Dc);
  k_attn<<<dim3(NWIN * NHc * 4 / 256), blk, 0, stream>>>(QKV, O);
  k_gemm<1><<<dim3(Dc / 128, Tc / 256), gblk, 0, stream>>>(
      O, Wo_bf, (void*)X2, bo, x, Tc, Dc, Dc);

  // Stage 2
  k_ln2<<<dim3(NWIN), blk, 0, stream>>>(X2, preg, preb, Mln);
  k_res<<<dim3((unsigned)(((long)NWIN * Dc + 255) / 256)), blk, 0, stream>>>(X2, RES);
  k_gemm<2><<<dim3((I4 + 127) / 128, NWIN / 256), gblk, 0, stream>>>(
      Mln, W1_bf, (void*)H1, b1, nullptr, NWIN, I4, D4);
  k_gemm<3><<<dim3(Dc / 128, NWIN / 256), gblk, 0, stream>>>(
      H1, W2_bf, d_out, b2, RES, NWIN, Dc, I4);
}
